// ComplexSelfAttention_43662637532044
// MI455X (gfx1250) — matrix operand order repaired and checked by reading, not compiled, not run on hardware
//
#include <hip/hip_runtime.h>
#include <hip/hip_bf16.h>

// ---------------------------------------------------------------------------
// ComplexSelfAttention on gfx1250 (MI455X): BN -> complex 1x1 QKV (WMMA GEMM)
// -> flash-style attention (WMMA QK^T + PV, online softmax) -> residual.
// All matmuls: v_wmma_f32_16x16x32_f16 (f16 in, f32 accumulate).
// ---------------------------------------------------------------------------

typedef __attribute__((ext_vector_type(16))) _Float16 v16h;
typedef __attribute__((ext_vector_type(8)))  float    v8f;

#define CB   128
#define NB   4096
#define BB   2
#define KDIM 256          // 2*C interleaved real/imag features

__device__ inline v8f vzero8() { v8f z = {}; return z; }

__device__ inline v8f wmma_f16(v16h a, v16h b, v8f c) {
  return __builtin_amdgcn_wmma_f32_16x16x32_f16(
      /*neg_a=*/false, a, /*neg_b=*/false, b,
      /*c_mod=*/(short)0, c, /*reuse_a=*/false, /*reuse_b=*/false);
}

// A-fragment (16x32 f16, M x K), source row-major with leading dim ld.
// Layout: lanes 0-15 row m hold K {hk..hk+15}, hk=0;
// lanes 16-31 same rows with hk=16.
__device__ inline v16h ld_afrag(const _Float16* A, int ld, int row0, int col0) {
  const int lane = threadIdx.x & 31;
  const int m  = row0 + (lane & 15);
  const int hk = (lane < 16) ? 0 : 16;
  const _Float16* p = A + (size_t)m * ld + col0;
  v16h a;
#pragma unroll
  for (int e = 0; e < 16; ++e) a[e] = p[hk + e];
  return a;
}

// B-fragment (32x16 f16, K x N) read from TRANSPOSED storage Bt[n][k]
// (leading dim ld over n rows). Lane holds column n = col0+(lane&15),
// K-contiguous span of 16 starting at k0 (+16 for upper half-wave).
__device__ inline v16h ld_bfrag_t(const _Float16* Bt, int ld, int col0, int k0) {
  const int lane = threadIdx.x & 31;
  const int n  = col0 + (lane & 15);
  const int kb = k0 + ((lane < 16) ? 0 : 16);
  const _Float16* p = Bt + (size_t)n * ld + kb;
  v16h b;
#pragma unroll
  for (int e = 0; e < 16; ++e) b[e] = p[e];
  return b;
}

__device__ inline void lds_fence() {
  __builtin_amdgcn_wave_barrier();
  asm volatile("s_wait_dscnt 0" ::: "memory");
  __builtin_amdgcn_wave_barrier();
}

// ------------------------- Kernel 1: BN statistics -------------------------
__global__ void bn_stats_kernel(const float* __restrict__ x,
                                float* __restrict__ stats) {
  const int cc = blockIdx.x;           // c*2 + comp, 0..255
  const int c = cc >> 1, comp = cc & 1;
  float s = 0.f, s2 = 0.f;
  for (int i = threadIdx.x; i < BB * NB; i += blockDim.x) {
    const int b = i >> 12, n = i & (NB - 1);
    const float v = x[(((size_t)(b * CB + c)) * NB + n) * 2 + comp];
    s += v; s2 += v * v;
  }
  __shared__ float sh0[256], sh1[256];
  sh0[threadIdx.x] = s; sh1[threadIdx.x] = s2;
  __syncthreads();
  for (int st = 128; st > 0; st >>= 1) {
    if ((int)threadIdx.x < st) {
      sh0[threadIdx.x] += sh0[threadIdx.x + st];
      sh1[threadIdx.x] += sh1[threadIdx.x + st];
    }
    __syncthreads();
  }
  if (threadIdx.x == 0) {
    const float inv = 1.0f / (float)(BB * NB);
    const float mean = sh0[0] * inv;
    const float var  = sh1[0] * inv - mean * mean;
    stats[cc]       = mean;
    stats[256 + cc] = rsqrtf(var + 1e-5f);
  }
}

// --------------- Kernel 2: BN apply -> xn (f32) + xcat (f16) ---------------
__global__ void bn_apply_kernel(const float* __restrict__ x,
                                const float* __restrict__ stats,
                                const float* __restrict__ bn_w,
                                const float* __restrict__ bn_b,
                                float* __restrict__ xn,
                                _Float16* __restrict__ xcat) {
  const int idx = blockIdx.x * blockDim.x + threadIdx.x;   // over B*C*N*2
  if (idx >= BB * CB * NB * 2) return;
  const int comp = idx & 1;
  const int n = (idx >> 1) & (NB - 1);
  const int c = (idx >> 13) & (CB - 1);
  const int b = idx >> 20;
  const int cc = c * 2 + comp;
  const float v = (x[idx] - stats[cc]) * stats[256 + cc] * bn_w[cc] + bn_b[cc];
  xn[idx] = v;
  xcat[((size_t)(b * NB + n)) * KDIM + cc] = (_Float16)v;   // token-major
}

// ----- Kernel 3: pack complex conv weights into real 256x256 f16 matrix -----
// row 2o   (real out): [2c]=wr, [2c+1]=-wi
// row 2o+1 (imag out): [2c]=wi, [2c+1]= wr
__global__ void pack_w_kernel(const float* __restrict__ w,
                              const float* __restrict__ bias,
                              _Float16* __restrict__ Wp,
                              float* __restrict__ bp) {
  const int idx = blockIdx.x * blockDim.x + threadIdx.x;   // 256*256
  if (idx >= KDIM * KDIM) return;
  const int row = idx >> 8, col = idx & 255;
  const int o = row >> 1, oc = row & 1;
  const int c = col >> 1, ci = col & 1;
  const float wr = w[((size_t)o * CB + c) * 2 + 0];
  const float wi = w[((size_t)o * CB + c) * 2 + 1];
  float val;
  if (oc == 0) val = (ci == 0) ? wr : -wi;
  else         val = (ci == 0) ? wi :  wr;
  Wp[idx] = (_Float16)val;
  if (col == 0) bp[row] = bias[o * 2 + oc];
}

// -------- Kernel 4: projection GEMM  out(256 x 4096) = Wp * xcat^T ---------
// Each wave: 16 out-channels x 64 tokens, K=256 in 8 WMMA steps.
__global__ void proj_gemm_kernel(const _Float16* __restrict__ Wp,
                                 const float* __restrict__ bp,
                                 const _Float16* __restrict__ xcat,
                                 _Float16* __restrict__ out,
                                 int token_major) {
  const int nt4 = blockIdx.x;          // 0..63 : token strip of 64
  const int mt  = blockIdx.y;          // 0..15 : out-channel tile
  const int b   = blockIdx.z;          // batch
  const _Float16* X = xcat + (size_t)b * NB * KDIM;
  const int lane = threadIdx.x & 31;

  v8f acc[4];
#pragma unroll
  for (int t = 0; t < 4; ++t) acc[t] = vzero8();

  for (int kk = 0; kk < 8; ++kk) {
    const v16h a = ld_afrag(Wp, KDIM, mt * 16, kk * 32);
#pragma unroll
    for (int t = 0; t < 4; ++t) {
      const v16h bf = ld_bfrag_t(X, KDIM, nt4 * 64 + t * 16, kk * 32);
      acc[t] = wmma_f16(a, bf, acc[t]);
    }
  }

  const int colL  = lane & 15;
  const int rbase = (lane < 16) ? 0 : 8;
#pragma unroll
  for (int t = 0; t < 4; ++t) {
    const int n = nt4 * 64 + t * 16 + colL;
#pragma unroll
    for (int v = 0; v < 8; ++v) {
      const int och = mt * 16 + rbase + v;
      const float val = acc[t][v] + bp[och];
      if (token_major)
        out[((size_t)b * NB + n) * KDIM + och] = (_Float16)val;   // [b][n][och]
      else
        out[((size_t)b * KDIM + och) * NB + n] = (_Float16)val;   // [b][och][n]
    }
  }
}

// ------------- Kernel 5: flash attention + residual epilogue ---------------
// One wave per 16-query tile. S = Q K^T (K-dim 256), online softmax,
// O += P V  (P routed through LDS to convert C-layout -> A-layout).
__global__ void flash_attn_kernel(const _Float16* __restrict__ qcat,
                                  const _Float16* __restrict__ kcat,
                                  const _Float16* __restrict__ vT,
                                  const float* __restrict__ xn,
                                  const float* __restrict__ gamma_p,
                                  float* __restrict__ out) {
  const int qt = blockIdx.x;           // 0..255
  const int b  = blockIdx.y;
  const int lane = threadIdx.x & 31;
  const _Float16* Q = qcat + (size_t)b * NB * KDIM;
  const _Float16* K = kcat + (size_t)b * NB * KDIM;
  const _Float16* V = vT   + (size_t)b * KDIM * NB;

  __shared__ _Float16 Pld[16 * 32];    // P tile, row-major 16x32

  v16h qf[8];
#pragma unroll
  for (int kk = 0; kk < 8; ++kk) qf[kk] = ld_afrag(Q, KDIM, qt * 16, kk * 32);

  float mrow[8], lrow[8];
#pragma unroll
  for (int v = 0; v < 8; ++v) { mrow[v] = -1e30f; lrow[v] = 0.f; }
  v8f O[16];
#pragma unroll
  for (int t = 0; t < 16; ++t) O[t] = vzero8();

  const float scale = 0.08838834764831845f;   // C^-0.5
  const int r0 = (lane < 16) ? 0 : 8;
  const int cl = lane & 15;

  for (int j = 0; j < 256; j += 2) {           // pairs of 16-key tiles
    if (j + 2 < 256) {                         // pull next K/V tiles toward L2
      __builtin_prefetch(K + (size_t)((j + 2) * 16) * KDIM, 0, 1);
      __builtin_prefetch(V + (size_t)(j + 2) * 16, 0, 1);
    }
    v8f S0 = vzero8(), S1 = vzero8();
#pragma unroll
    for (int kk = 0; kk < 8; ++kk) {
      const v16h b0 = ld_bfrag_t(K, KDIM, (j + 0) * 16, kk * 32);
      S0 = wmma_f16(qf[kk], b0, S0);
      const v16h b1 = ld_bfrag_t(K, KDIM, (j + 1) * 16, kk * 32);
      S1 = wmma_f16(qf[kk], b1, S1);
    }

    float alpha[8];
#pragma unroll
    for (int v = 0; v < 8; ++v) {
      S0[v] *= scale; S1[v] *= scale;
      float mx = fmaxf(S0[v], S1[v]);          // row reduce over 16 lanes
      mx = fmaxf(mx, __shfl_xor(mx, 1, 32));
      mx = fmaxf(mx, __shfl_xor(mx, 2, 32));
      mx = fmaxf(mx, __shfl_xor(mx, 4, 32));
      mx = fmaxf(mx, __shfl_xor(mx, 8, 32));
      const float newm = fmaxf(mrow[v], mx);
      alpha[v] = __expf(mrow[v] - newm);
      mrow[v]  = newm;
      const float p0 = __expf(S0[v] - newm);
      const float p1 = __expf(S1[v] - newm);
      S0[v] = p0; S1[v] = p1;
      float rs = p0 + p1;
      rs += __shfl_xor(rs, 1, 32);
      rs += __shfl_xor(rs, 2, 32);
      rs += __shfl_xor(rs, 4, 32);
      rs += __shfl_xor(rs, 8, 32);
      lrow[v] = lrow[v] * alpha[v] + rs;
    }
#pragma unroll
    for (int t = 0; t < 16; ++t)
#pragma unroll
      for (int v = 0; v < 8; ++v) O[t][v] *= alpha[v];

    // C-layout -> A-layout through LDS (16 rows x 32 keys).
#pragma unroll
    for (int v = 0; v < 8; ++v) {
      Pld[(v + r0) * 32 + cl]      = (_Float16)S0[v];
      Pld[(v + r0) * 32 + 16 + cl] = (_Float16)S1[v];
    }
    lds_fence();
    const v16h pf = ld_afrag(Pld, 32, 0, 0);

#pragma unroll
    for (int t = 0; t < 16; ++t) {
      // B[k=key][n=channel]: vT rows are channel-major, K-contiguous.
      const v16h bv = ld_bfrag_t(V, NB, t * 16, j * 16);
      O[t] = wmma_f16(pf, bv, O[t]);
    }
  }

  const float g = gamma_p[0];
#pragma unroll
  for (int t = 0; t < 16; ++t) {
    const int kch = t * 16 + cl;
    const int c = kch >> 1, comp = kch & 1;
#pragma unroll
    for (int v = 0; v < 8; ++v) {
      const int n = qt * 16 + r0 + v;
      const size_t idx = (((size_t)(b * CB + c)) * NB + n) * 2 + comp;
      out[idx] = xn[idx] + g * (O[t][v] / lrow[v]);
    }
  }
}

// ---------------------------------------------------------------------------
extern "C" void kernel_launch(void* const* d_in, const int* in_sizes, int n_in,
                              void* d_out, int out_size, void* d_ws, size_t ws_size,
                              hipStream_t stream) {
  const float* x    = (const float*)d_in[0];
  const float* bn_w = (const float*)d_in[1];
  const float* bn_b = (const float*)d_in[2];
  const float* wq   = (const float*)d_in[3];
  const float* bq   = (const float*)d_in[4];
  const float* wk   = (const float*)d_in[5];
  const float* bk   = (const float*)d_in[6];
  const float* wv   = (const float*)d_in[7];
  const float* bv   = (const float*)d_in[8];
  const float* gm   = (const float*)d_in[9];
  float* out = (float*)d_out;

  char* ws = (char*)d_ws;
  // workspace layout (bytes)
  float*    stats = (float*)(ws + 0x0);                        // 512 f32
  float*    xn    = (float*)(ws + 0x1000);                     // 8 MB f32
  _Float16* xcat  = (_Float16*)(ws + 0x801000);                // 4 MB f16
  _Float16* wp_q  = (_Float16*)(ws + 0xC01000);                // 128 KB each
  _Float16* wp_k  = wp_q + KDIM * KDIM;
  _Float16* wp_v  = wp_k + KDIM * KDIM;
  float*    bp_q  = (float*)(ws + 0xC61000);                   // 1 KB each
  float*    bp_k  = bp_q + KDIM;
  float*    bp_v  = bp_k + KDIM;
  _Float16* qcat  = (_Float16*)(ws + 0xC62000);                // 4 MB
  _Float16* kcat  = (_Float16*)(ws + 0x1062000);               // 4 MB
  _Float16* vT    = (_Float16*)(ws + 0x1462000);               // 4 MB

  const int total = BB * CB * NB * 2;

  bn_stats_kernel<<<256, 256, 0, stream>>>(x, stats);
  bn_apply_kernel<<<(total + 255) / 256, 256, 0, stream>>>(x, stats, bn_w, bn_b,
                                                           xn, xcat);
  pack_w_kernel<<<(KDIM * KDIM + 255) / 256, 256, 0, stream>>>(wq, bq, wp_q, bp_q);
  pack_w_kernel<<<(KDIM * KDIM + 255) / 256, 256, 0, stream>>>(wk, bk, wp_k, bp_k);
  pack_w_kernel<<<(KDIM * KDIM + 255) / 256, 256, 0, stream>>>(wv, bv, wp_v, bp_v);

  dim3 pgrid(64, 16, BB);
  proj_gemm_kernel<<<pgrid, 32, 0, stream>>>(wp_q, bp_q, xcat, qcat, 1);
  proj_gemm_kernel<<<pgrid, 32, 0, stream>>>(wp_k, bp_k, xcat, kcat, 1);
  proj_gemm_kernel<<<pgrid, 32, 0, stream>>>(wp_v, bp_v, xcat, vT,   0);

  dim3 agrid(NB / 16, BB);
  flash_attn_kernel<<<agrid, 32, 0, stream>>>(qcat, kcat, vT, xn, gm, out);
}
